// GraphSkip_48163763257697
// MI455X (gfx1250) — matrix operand order repaired and checked by reading, not compiled, not run on hardware
//
#include <hip/hip_runtime.h>
#include <hip/hip_bf16.h>
#include <stdint.h>

// ---------------------------------------------------------------------------
// GraphSAGE x3 with skip connections, N=100000, E=1600000, F=128.
// Scatter-aggregate with f32 atomics (memory bound, dominates);
// dense math via v_wmma_f32_16x16x32_bf16. Register-pressure-aware GEMM:
// 8 v8f accumulators per wave, one live A fragment, WMMAs interleaved
// across independent accumulators to avoid D->C stalls and spills.
// ---------------------------------------------------------------------------

typedef __attribute__((ext_vector_type(16))) __bf16 v16bf;
typedef __attribute__((ext_vector_type(8)))  float  v8f;

#define FT 128          // feature width (in == out == 128)
#define KSTEPS 4        // 128 / 32
#define NTILES 8        // 128 / 16
#define FRAG_ELEMS 512  // 32 lanes * 16 bf16 per B fragment
#define W_ELEMS (KSTEPS * NTILES * FRAG_ELEMS) // 16384 bf16 per packed weight

// --------------------------- weight pre-pack -------------------------------
// Pack a 128x128 row-major f32 weight into bf16 WMMA-B fragment layout:
// fragment(ks,nt) is 32 lanes x 16 contiguous bf16 (32B per lane, coalesced).
// B layout (16-bit, 32x16): lane group g holds K = j + 8g (j<8) or j + 8 + 8g (j>=8), column N = lane%16.
__global__ void pack_weights_kernel(const float* __restrict__ w0,
                                    const float* __restrict__ w1,
                                    const float* __restrict__ w2,
                                    const float* __restrict__ w3,
                                    const float* __restrict__ w4,
                                    const float* __restrict__ w5,
                                    const float* __restrict__ w6,
                                    __bf16* __restrict__ outBase) {
    const float* ws[7] = {w0, w1, w2, w3, w4, w5, w6};
    const float* W = ws[blockIdx.x];
    __bf16* o = outBase + (size_t)blockIdx.x * W_ELEMS;
    for (int idx = threadIdx.x; idx < W_ELEMS; idx += blockDim.x) {
        int j    = idx & 15;
        int lane = (idx >> 4) & 31;
        int frag = idx >> 9;          // ks*8 + nt
        int ks   = frag >> 3;
        int nt   = frag & 7;
        int l    = lane & 15;
        int g    = lane >> 4;
        int k    = ks * 32 + g * 8 + j + ((j >> 3) << 3);   // B-matrix 16-bit layout
        int n    = nt * 16 + l;
        o[idx] = (__bf16)W[k * FT + n];
    }
}

// ----------------------------- degree --------------------------------------
__global__ void deg_kernel(const int* __restrict__ dst, float* __restrict__ deg, int E) {
    int e = blockIdx.x * blockDim.x + threadIdx.x;
    if (e < E) atomicAdd(&deg[dst[e]], 1.0f);
}

__global__ void invdeg_kernel(float* __restrict__ deg, int N) {
    int i = blockIdx.x * blockDim.x + threadIdx.x;
    if (i < N) deg[i] = 1.0f / fmaxf(deg[i], 1.0f);
}

// ----------------------------- scatter --------------------------------------
// One wave32 per edge: 128 floats -> 4 coalesced f32 atomics per lane.
__global__ void scatter_kernel(const float* __restrict__ feat,
                               const int* __restrict__ src,
                               const int* __restrict__ dst,
                               float* __restrict__ agg, int E) {
    int wid  = (int)((blockIdx.x * blockDim.x + threadIdx.x) >> 5);
    int lane = threadIdx.x & 31;
    if (wid >= E) return;
    int s = src[wid];
    int d = dst[wid];
    const float* fs = feat + (size_t)s * FT;
    float*       fd = agg  + (size_t)d * FT;
#pragma unroll
    for (int i = 0; i < 4; ++i)
        atomicAdd(&fd[lane + 32 * i], fs[lane + 32 * i]);
}

// A-fragment builder. A layout (16-bit, 16x32): lane group g holds
// K = j + 8g (j<8) or j + 8 + 8g (j>=8). `scale` multiplies before cvt.
__device__ __forceinline__ v16bf make_a_frag(const float* __restrict__ rowp,
                                             int kbase, float scale) {
    v16bf a;
#pragma unroll
    for (int j = 0; j < 16; ++j) {
        int k = kbase + j + ((j >> 3) << 3);
        a[j] = (__bf16)(rowp[k] * scale);
    }
    return a;
}

// --------------------------- plain skip GEMM --------------------------------
// out = act @ W  (no bias, no activation). One wave per 16-row panel.
__global__ void skip_gemm_kernel(const float* __restrict__ act,
                                 const __bf16* __restrict__ pW,
                                 float* __restrict__ out, int nrows) {
    const int lane  = threadIdx.x & 31;
    const int wib   = threadIdx.x >> 5;
    const int panel = blockIdx.x * (blockDim.x >> 5) + wib;
    const int m0    = panel * 16;
    if (m0 >= nrows) return;             // wave-uniform

    const int l = lane & 15;
    const int g = lane >> 4;
    const float* rowp = act + (size_t)(m0 + l) * FT;

    v8f acc[NTILES];
#pragma unroll
    for (int i = 0; i < NTILES; ++i) acc[i] = (v8f)0.0f;

#pragma unroll
    for (int ks = 0; ks < KSTEPS; ++ks) {
        v16bf a = make_a_frag(rowp, ks * 32 + g * 8, 1.0f);
        const __bf16* bp = pW + ks * NTILES * FRAG_ELEMS + lane * 16;
#pragma unroll
        for (int nt = 0; nt < NTILES; ++nt) {
            v16bf b = *(const v16bf*)(bp + nt * FRAG_ELEMS);
            acc[nt] = __builtin_amdgcn_wmma_f32_16x16x32_bf16(
                false, a, false, b, (short)0, acc[nt], false, false);
        }
    }
#pragma unroll
    for (int nt = 0; nt < NTILES; ++nt) {
        const int col = nt * 16 + l;
#pragma unroll
        for (int r = 0; r < 8; ++r)
            out[(size_t)(m0 + r + 8 * g) * FT + col] = acc[nt][r];
    }
}

// --------------------------- fused SAGE GEMM --------------------------------
// One wave per 16-row panel:
//   h   = prelu( (agg*inv_deg) @ Wl + bl + act @ Wr )
//   out = (res ? res + h : h)
// Both GEMMs share the 8 accumulators; only one A fragment is live at a time,
// and WMMAs sweep across the 8 independent n-tile accumulators.
__global__ void sage_gemm_kernel(const float* __restrict__ act,
                                 const float* __restrict__ agg,
                                 const float* __restrict__ inv_deg,
                                 const __bf16* __restrict__ pWl,
                                 const float* __restrict__ bl,
                                 const __bf16* __restrict__ pWr,
                                 const float* __restrict__ res,   // may be null
                                 const float* __restrict__ prelu_a,
                                 float* __restrict__ out,
                                 int nrows) {
    const int lane  = threadIdx.x & 31;
    const int wib   = threadIdx.x >> 5;
    const int panel = blockIdx.x * (blockDim.x >> 5) + wib;
    const int m0    = panel * 16;
    if (m0 >= nrows) return;             // wave-uniform: EXEC stays all-ones

    const int l = lane & 15;
    const int g = lane >> 4;
    const int row = m0 + l;
    const float inv = inv_deg[row];
    const float pa  = *prelu_a;

    const float* actRow = act + (size_t)row * FT;
    const float* aggRow = agg + (size_t)row * FT;

    v8f acc[NTILES];
#pragma unroll
    for (int i = 0; i < NTILES; ++i) acc[i] = (v8f)0.0f;

#pragma unroll
    for (int ks = 0; ks < KSTEPS; ++ks) {
        const int kbase = ks * 32 + g * 8;
        const __bf16* blf = pWl + ks * NTILES * FRAG_ELEMS + lane * 16;
        const __bf16* brf = pWr + ks * NTILES * FRAG_ELEMS + lane * 16;
        {
            v16bf aAgg = make_a_frag(aggRow, kbase, inv);
#pragma unroll
            for (int nt = 0; nt < NTILES; ++nt) {
                v16bf b = *(const v16bf*)(blf + nt * FRAG_ELEMS);
                acc[nt] = __builtin_amdgcn_wmma_f32_16x16x32_bf16(
                    false, aAgg, false, b, (short)0, acc[nt], false, false);
            }
        }
        {
            v16bf aAct = make_a_frag(actRow, kbase, 1.0f);
#pragma unroll
            for (int nt = 0; nt < NTILES; ++nt) {
                v16bf b = *(const v16bf*)(brf + nt * FRAG_ELEMS);
                acc[nt] = __builtin_amdgcn_wmma_f32_16x16x32_bf16(
                    false, aAct, false, b, (short)0, acc[nt], false, false);
            }
        }
    }

    // Epilogue. C/D layout: element (M = r + 8g, N = l) in acc[r].
    const bool hasRes = (res != nullptr);  // wave-uniform
#pragma unroll
    for (int nt = 0; nt < NTILES; ++nt) {
        const int col = nt * 16 + l;
        const float b = bl[col];
#pragma unroll
        for (int r = 0; r < 8; ++r) {
            const size_t idx = (size_t)(m0 + r + 8 * g) * FT + col;
            float h = acc[nt][r] + b;
            h = (h >= 0.0f) ? h : pa * h;
            out[idx] = hasRes ? (res[idx] + h) : h;
        }
    }
}

// ------------------------------- launcher -----------------------------------
extern "C" void kernel_launch(void* const* d_in, const int* in_sizes, int n_in,
                              void* d_out, int out_size, void* d_ws, size_t ws_size,
                              hipStream_t stream) {
    const float* x     = (const float*)d_in[0];
    const int*   edges = (const int*)d_in[1];
    const float* Wskip = (const float*)d_in[2];
    const float* Wl1   = (const float*)d_in[3];
    const float* bl1   = (const float*)d_in[4];
    const float* Wr1   = (const float*)d_in[5];
    const float* Wl2   = (const float*)d_in[6];
    const float* bl2   = (const float*)d_in[7];
    const float* Wr2   = (const float*)d_in[8];
    const float* Wl3   = (const float*)d_in[9];
    const float* bl3   = (const float*)d_in[10];
    const float* Wr3   = (const float*)d_in[11];
    const float* pa    = (const float*)d_in[12];
    float* out = (float*)d_out;

    const int N = in_sizes[0] / FT;
    const int E = in_sizes[1] / 2;
    const int* src = edges;
    const int* dst = edges + E;

    // Workspace layout (256B aligned segments)
    char* ws = (char*)d_ws;
    size_t off = 0;
    auto take = [&](size_t bytes) -> char* {
        char* p = ws + off;
        off = (off + bytes + 255) & ~(size_t)255;
        return p;
    };
    __bf16* packW = (__bf16*)take((size_t)7 * W_ELEMS * sizeof(__bf16));
    float*  deg   = (float*)take((size_t)N * sizeof(float));           // -> inv_deg
    float*  aggb  = (float*)take((size_t)N * FT * sizeof(float));
    float*  z1    = (float*)take((size_t)N * FT * sizeof(float));

    // Packed weight slots: 0=Wskip 1=Wl1 2=Wr1 3=Wl2 4=Wr2 5=Wl3 6=Wr3
    __bf16* pWskip = packW + 0 * W_ELEMS;
    __bf16* pWl1   = packW + 1 * W_ELEMS;
    __bf16* pWr1   = packW + 2 * W_ELEMS;
    __bf16* pWl2   = packW + 3 * W_ELEMS;
    __bf16* pWr2   = packW + 4 * W_ELEMS;
    __bf16* pWl3   = packW + 5 * W_ELEMS;
    __bf16* pWr3   = packW + 6 * W_ELEMS;

    const size_t aggBytes = (size_t)N * FT * sizeof(float);
    const int panels    = (N + 15) / 16;
    const int gemmGrid  = (panels + 7) / 8;       // 8 waves (256 thr) per block
    const int edgeGridT = (E + 255) / 256;        // thread-per-edge kernels
    const int edgeGridW = (E + 7) / 8;            // wave-per-edge kernels

    pack_weights_kernel<<<7, 256, 0, stream>>>(Wskip, Wl1, Wr1, Wl2, Wr2, Wl3, Wr3, packW);

    // degree -> inv_deg
    hipMemsetAsync(deg, 0, (size_t)N * sizeof(float), stream);
    deg_kernel<<<edgeGridT, 256, 0, stream>>>(dst, deg, E);
    invdeg_kernel<<<(N + 255) / 256, 256, 0, stream>>>(deg, N);

    // s = x @ Wskip  (stored in z1; layer-1 SAGE adds prelu(...) in place)
    skip_gemm_kernel<<<gemmGrid, 256, 0, stream>>>(x, pWskip, z1, N);

    // ---- layer 1: z1 = s + prelu(mean(x) @ Wl1 + bl1 + x @ Wr1)
    hipMemsetAsync(aggb, 0, aggBytes, stream);
    scatter_kernel<<<edgeGridW, 256, 0, stream>>>(x, src, dst, aggb, E);
    sage_gemm_kernel<<<gemmGrid, 256, 0, stream>>>(
        x, aggb, deg, pWl1, bl1, pWr1, /*res=*/z1, pa, z1, N);

    // ---- layer 2: z2 = z1 + prelu(mean(z1) @ Wl2 + bl2 + z1 @ Wr2)  (z2 in d_out)
    hipMemsetAsync(aggb, 0, aggBytes, stream);
    scatter_kernel<<<edgeGridW, 256, 0, stream>>>(z1, src, dst, aggb, E);
    sage_gemm_kernel<<<gemmGrid, 256, 0, stream>>>(
        z1, aggb, deg, pWl2, bl2, pWr2, /*res=*/z1, pa, out, N);

    // ---- layer 3: h3 = prelu(mean(z2) @ Wl3 + bl3 + z2 @ Wr3)  (in-place on d_out)
    hipMemsetAsync(aggb, 0, aggBytes, stream);
    scatter_kernel<<<edgeGridW, 256, 0, stream>>>(out, src, dst, aggb, E);
    sage_gemm_kernel<<<gemmGrid, 256, 0, stream>>>(
        out, aggb, deg, pWl3, bl3, pWr3, /*res=*/nullptr, pa, out, N);
}
